// GENELink_15496242004384
// MI455X (gfx1250) — compile-verified, matrix operand order checked by reading, not run on hardware
//
#include <hip/hip_runtime.h>
#include <hip/hip_bf16.h>
#include <math.h>

// ---------------------------------------------------------------- types
typedef __attribute__((ext_vector_type(16))) __bf16 v16bf;
typedef __attribute__((ext_vector_type(8)))  float  v8f;

union Frag { v16bf b; uint4 q[2]; };

__device__ __forceinline__ float lrelu(float x, float s) { return x > 0.f ? x : x * s; }

#define GN 4096     // nodes
#define ALPHA 0.2f
#define SLOPE 0.01f

// ---------------------------------------------------------------- operand pre-pack
// Pack a KxN f32 matrix into WMMA B-fragment order (bf16):
//   Bp[tile*512 + lane*16 + e] = bf16( B[kt*32 + klocal(e,lane), tn*16 + (lane&15)] )
// with tile = kt*(N/16)+tn, klocal = (e&7) + 8*(lane>>4) + ((e&8)?16:0).
// Consumers then fetch a full fragment as two contiguous b128 loads per lane.
__global__ void pack_b(const float* __restrict__ B, __bf16* __restrict__ Bp, int K, int N) {
    const int idx = blockIdx.x * 256 + threadIdx.x;
    if (idx >= K * N) return;
    const int e      = idx & 15;
    const int lane   = (idx >> 4) & 31;
    const int tile   = idx >> 9;
    const int ntiles = N >> 4;
    const int tn     = tile % ntiles;
    const int kt     = tile / ntiles;
    const int khalf  = lane >> 4, nn = lane & 15;
    const int kl     = (e & 7) + (khalf << 3) + ((e & 8) ? 16 : 0);
    Bp[idx] = (__bf16)B[(size_t)((kt << 5) + kl) * N + ((tn << 4) + nn)];
}

// ---------------------------------------------------------------- generic bf16-WMMA GEMM
// One wave per 16x16 C tile.  C = act(A(MxK,f32) @ Bp(pre-packed bf16) + bias).
// A lane's 16 A elements are two contiguous 8-float K-groups -> 4x float4 loads + cvt.
template <int ACT>
__global__ __launch_bounds__(32)
void wmma_gemm(const float* __restrict__ A, const __bf16* __restrict__ Bp,
               const float* __restrict__ bias, float* __restrict__ C,
               int M, int N, int K, float slope) {
    const int tm = blockIdx.x, tn = blockIdx.y;
    const int lane  = threadIdx.x;
    const int nn    = lane & 15;
    const int khalf = lane >> 4;
    const int ntiles = N >> 4;

    v8f acc = {};
    for (int k0 = 0; k0 < K; k0 += 32) {
        const float* ar = A + (size_t)(tm * 16 + nn) * K + k0 + (khalf << 3);
        const float4 a0 = *(const float4*)(ar);
        const float4 a1 = *(const float4*)(ar + 4);
        const float4 a2 = *(const float4*)(ar + 16);
        const float4 a3 = *(const float4*)(ar + 20);
        const float tmp[16] = { a0.x, a0.y, a0.z, a0.w, a1.x, a1.y, a1.z, a1.w,
                                a2.x, a2.y, a2.z, a2.w, a3.x, a3.y, a3.z, a3.w };
        Frag fa, fb;
#pragma unroll
        for (int e = 0; e < 16; ++e) fa.b[e] = (__bf16)tmp[e];

        const uint4* bq = (const uint4*)(Bp + (((size_t)(k0 >> 5) * ntiles + tn) << 9)) + lane * 2;
        fb.q[0] = bq[0];
        fb.q[1] = bq[1];

        acc = __builtin_amdgcn_wmma_f32_16x16x32_bf16(
            false, fa.b, false, fb.b, (short)0, acc, false, false);
    }

    const int col = tn * 16 + nn;
    const float bv = bias ? bias[col] : 0.f;
#pragma unroll
    for (int v = 0; v < 8; ++v) {
        const int row = tm * 16 + v + 8 * khalf;
        float val = acc[v] + bv;
        if (ACT == 1) val = lrelu(val, slope);
        C[(size_t)row * N + col] = val;
    }
}

// ---------------------------------------------------------------- s1/s2 = h @ a_lo , h @ a_hi
__global__ __launch_bounds__(32)
void row_dots(const float* __restrict__ h, const float* __restrict__ a,
              int F, float* __restrict__ s1, float* __restrict__ s2) {
    const int r = blockIdx.x, lane = threadIdx.x;
    const float* hr = h + (size_t)r * F;
    float a1 = 0.f, a2 = 0.f;
    for (int f = lane; f < F; f += 32) {
        const float v = hr[f];
        a1 += v * a[f];
        a2 += v * a[F + f];
    }
    for (int off = 16; off; off >>= 1) { a1 += __shfl_xor(a1, off); a2 += __shfl_xor(a2, off); }
    if (lane == 0) { s1[r] = a1; s2[r] = a2; }
}

// ---------------------------------------------------------------- per-row masked max of s2 (3 heads at once)
__global__ __launch_bounds__(256)
void row_neigh_max(const int* __restrict__ adj, const float* __restrict__ s2 /*3xN*/,
                   int N, float* __restrict__ m /*3xN*/) {
    const int i = blockIdx.x, t = threadIdx.x;
    float mx0 = -__builtin_inff(), mx1 = mx0, mx2 = mx0;
    const int* arow = adj + (size_t)i * N;
    for (int j = t; j < N; j += 256) {
        if (arow[j] > 0) {
            mx0 = fmaxf(mx0, s2[j]);
            mx1 = fmaxf(mx1, s2[N + j]);
            mx2 = fmaxf(mx2, s2[2 * N + j]);
        }
    }
    __shared__ float red[3][256];
    red[0][t] = mx0; red[1][t] = mx1; red[2][t] = mx2;
    __syncthreads();
    for (int off = 128; off; off >>= 1) {
        if (t < off) {
            red[0][t] = fmaxf(red[0][t], red[0][t + off]);
            red[1][t] = fmaxf(red[1][t], red[1][t + off]);
            red[2][t] = fmaxf(red[2][t], red[2][t + off]);
        }
        __syncthreads();
    }
    if (t == 0) { m[i] = red[0][0]; m[N + i] = red[1][0]; m[2 * N + i] = red[2][0]; }
}

// ---------------------------------------------------------------- fused flash-style GAT head
// out = leaky( softmax_row(mask(leaky(s1_i+s2_j))) @ h ) row-L2-normalized + bias.
// Block: 4 waves (128 thr), 16 rows, full FF features.  Exact row maxima precomputed,
// so a single pass accumulates unnormalized O and Z.  P tile is written straight into
// A-fragment order in LDS (consumers read 2x ds_load_b128); h comes pre-packed in
// B-fragment order (2x global b128 per tile).  FF is a compile-time constant so the
// per-wave accumulators stay in statically-addressed VGPRs (no v_movrel indexing).
template <int FF>
__global__ __launch_bounds__(128)
void flash_head(const float* __restrict__ s1, const float* __restrict__ s2,
                const float* __restrict__ mrow, const int* __restrict__ adj,
                const __bf16* __restrict__ hp, const float* __restrict__ bias,
                float* __restrict__ out, int N) {
    constexpr int NTILE = FF >> 4;        // 16-wide feature tiles (8 or 4)
    constexpr int NACC  = FF >> 6;        // tiles per wave (2 or 1)

    __shared__ __align__(16) __bf16 plds[512];     // 16x32 P tile, fragment order
    __shared__ float zlds[16];
    __shared__ float normlds[16];

    const int t    = threadIdx.x;
    const int lane = t & 31;
    const int wave = t >> 5;
    const int i0   = blockIdx.x * 16;
    const int c    = lane;            // neighbor column within 32-wide j tile
    const int nn    = lane & 15;
    const int khalf = lane >> 4;

    if (t < 16) { zlds[t] = 0.f; normlds[t] = 0.f; }

    // destination slot in fragment-order LDS for this thread's P values at column c
    const int eslot = (c & 7) + 8 * (c >> 4);      // element index for k = c
    const int lhalf = (c >> 3) & 1;                // lane-half that owns k = c

    float s1v[4], mv[4], zacc[4];
#pragma unroll
    for (int q = 0; q < 4; ++q) {
        const int r = wave + 4 * q;
        s1v[q] = s1[i0 + r];
        mv[q]  = mrow[i0 + r];
        zacc[q] = 0.f;
    }

    v8f acc[NACC];
#pragma unroll
    for (int ti = 0; ti < NACC; ++ti) acc[ti] = (v8f){};

    for (int j0 = 0; j0 < N; j0 += 32) {
        const float s2c = s2[j0 + c];
        if (j0 + 32 < N)    // hide adj latency of the next tile (global_prefetch_b8)
            __builtin_prefetch(&adj[(size_t)(i0 + wave) * N + j0 + 32 + c], 0, 3);
#pragma unroll
        for (int q = 0; q < 4; ++q) {
            const int r = wave + 4 * q;
            const int av = adj[(size_t)(i0 + r) * N + j0 + c];
            float p = 0.f;
            if (av > 0) {
                float sc = s1v[q] + s2c;
                sc = sc > 0.f ? sc : sc * ALPHA;
                p = __expf(sc - mv[q]);
            }
            plds[(r + 16 * lhalf) * 16 + eslot] = (__bf16)p;
            zacc[q] += p;
        }
        __syncthreads();

        Frag fa;                                  // A frag: shared by this wave's tiles
        {
            const uint4* pq = (const uint4*)plds + lane * 2;
            fa.q[0] = pq[0];
            fa.q[1] = pq[1];
        }
#pragma unroll
        for (int ti = 0; ti < NACC; ++ti) {
            const int tt = wave + ti * 4;         // this wave's feature tile
            Frag fb;                              // pre-packed B frag: h rows j0..j0+31
            const uint4* hq =
                (const uint4*)(hp + (((size_t)(j0 >> 5) * NTILE + tt) << 9)) + lane * 2;
            fb.q[0] = hq[0];
            fb.q[1] = hq[1];
            acc[ti] = __builtin_amdgcn_wmma_f32_16x16x32_bf16(
                false, fa.b, false, fb.b, (short)0, acc[ti], false, false);
        }
        __syncthreads();
    }

    // reduce per-row Z across the wave's 32 neighbor columns
#pragma unroll
    for (int q = 0; q < 4; ++q) {
        float z = zacc[q];
        for (int off = 16; off; off >>= 1) z += __shfl_xor(z, off);
        if (lane == 0) zlds[wave + 4 * q] = z;
    }
    __syncthreads();

    // Z-normalize + leaky, accumulate row sum-of-squares
#pragma unroll
    for (int ti = 0; ti < NACC; ++ti) {
#pragma unroll
        for (int v = 0; v < 8; ++v) {
            const int row = v + 8 * khalf;
            float val = acc[ti][v] / fmaxf(zlds[row], 1e-30f);
            val = lrelu(val, ALPHA);
            acc[ti][v] = val;
            atomicAdd(&normlds[row], val * val);
        }
    }
    __syncthreads();

    // L2 normalize + bias, store
#pragma unroll
    for (int ti = 0; ti < NACC; ++ti) {
        const int tt  = wave + ti * 4;
        const int col = tt * 16 + nn;
        const float bv = bias[col];
#pragma unroll
        for (int v = 0; v < 8; ++v) {
            const int row = v + 8 * khalf;
            const float denom = fmaxf(sqrtf(normlds[row]), 1e-12f);
            out[(size_t)(i0 + row) * FF + col] = acc[ti][v] / denom + bv;
        }
    }
}

// ---------------------------------------------------------------- elementwise helpers
__global__ void elu_concat(const float* __restrict__ o /*3 x N x F*/,
                           float* __restrict__ hcat, int N, int F) {
    const int idx = blockIdx.x * 256 + threadIdx.x;
    const int total = N * F * 3;
    if (idx >= total) return;
    const int fh = idx % (F * 3);
    const int r  = idx / (F * 3);
    const int l  = fh / F;
    const int ff = fh % F;
    float v = o[((size_t)l * N + r) * F + ff];
    v = v > 0.f ? v : (__expf(v) - 1.f);
    hcat[idx] = v;
}

__global__ void mean3(const float* __restrict__ o /*3 x NF*/, float* __restrict__ dst, int NF) {
    const int i = blockIdx.x * 256 + threadIdx.x;
    if (i >= NF) return;
    dst[i] = (o[i] + o[NF + i] + o[2 * NF + i]) * (1.f / 3.f);
}

__global__ void pred_dot(const int* __restrict__ ts, const float* __restrict__ tf,
                         const float* __restrict__ tg, float* __restrict__ pred, int E) {
    const int p = blockIdx.x * 256 + threadIdx.x;
    if (p >= E) return;
    const int i = ts[2 * p], j = ts[2 * p + 1];
    const float4* a = (const float4*)(tf + (size_t)i * 16);
    const float4* b = (const float4*)(tg + (size_t)j * 16);
    float s = 0.f;
#pragma unroll
    for (int k = 0; k < 4; ++k) {
        const float4 av = a[k], bv = b[k];
        s += av.x * bv.x + av.y * bv.y + av.z * bv.z + av.w * bv.w;
    }
    pred[p] = s;
}

// ---------------------------------------------------------------- launch
extern "C" void kernel_launch(void* const* d_in, const int* in_sizes, int n_in,
                              void* d_out, int out_size, void* d_ws, size_t ws_size,
                              hipStream_t stream) {
    (void)in_sizes; (void)n_in; (void)out_size; (void)ws_size;
    const int N = GN, D = 512, F1 = 128, F2 = 64, H3 = 64, DOUT = 16, E = 131072;

    const float* x    = (const float*)d_in[0];
    const int*   adj  = (const int*)  d_in[1];
    const int*   ts   = (const int*)  d_in[2];
    const float* W1   = (const float*)d_in[3];   // (3, 512, 128)
    const float* a1   = (const float*)d_in[4];   // (3, 256, 1)
    const float* b1   = (const float*)d_in[5];   // (3, 128)
    const float* W2   = (const float*)d_in[6];   // (3, 384, 64)
    const float* a2   = (const float*)d_in[7];   // (3, 128, 1)
    const float* b2   = (const float*)d_in[8];   // (3, 64)
    const float* tfW1 = (const float*)d_in[9];
    const float* tfb1 = (const float*)d_in[10];
    const float* tfW2 = (const float*)d_in[11];
    const float* tfb2 = (const float*)d_in[12];
    const float* tgW1 = (const float*)d_in[13];
    const float* tgb1 = (const float*)d_in[14];
    const float* tgW2 = (const float*)d_in[15];
    const float* tgb2 = (const float*)d_in[16];

    // output partitions (return order: embed, tf, tg, pred)
    float* embed_out = (float*)d_out;
    float* tf_out    = embed_out + (size_t)N * F2;
    float* tg_out    = tf_out    + (size_t)N * DOUT;
    float* pred_out  = tg_out    + (size_t)N * DOUT;

    // workspace bump allocator (256B aligned)
    char* w = (char*)d_ws;
    auto alloc = [&](size_t bytes) -> void* {
        void* p = (void*)w;
        w += (bytes + 255) & ~(size_t)255;
        return p;
    };
    float*  h1    = (float*) alloc((size_t)3 * N * F1 * 4);
    __bf16* h1p   = (__bf16*)alloc((size_t)3 * N * F1 * 2);
    float*  o1    = (float*) alloc((size_t)3 * N * F1 * 4);
    float*  hcat  = (float*) alloc((size_t)N * 3 * F1 * 4);
    float*  h2    = (float*) alloc((size_t)3 * N * F2 * 4);
    __bf16* h2p   = (__bf16*)alloc((size_t)3 * N * F2 * 2);
    float*  o2    = (float*) alloc((size_t)3 * N * F2 * 4);
    float*  s1b   = (float*) alloc((size_t)3 * N * 4);
    float*  s2b   = (float*) alloc((size_t)3 * N * 4);
    float*  mb    = (float*) alloc((size_t)3 * N * 4);
    float*  s1b2  = (float*) alloc((size_t)3 * N * 4);
    float*  s2b2  = (float*) alloc((size_t)3 * N * 4);
    float*  mb2   = (float*) alloc((size_t)3 * N * 4);
    float*  tf1   = (float*) alloc((size_t)N * H3 * 4);
    float*  tg1   = (float*) alloc((size_t)N * H3 * 4);
    __bf16* W1p   = (__bf16*)alloc((size_t)3 * D * F1 * 2);
    __bf16* W2p   = (__bf16*)alloc((size_t)3 * (3 * F1) * F2 * 2);
    __bf16* tfW1p = (__bf16*)alloc((size_t)F2 * H3 * 2);
    __bf16* tfW2p = (__bf16*)alloc((size_t)H3 * DOUT * 2);
    __bf16* tgW1p = (__bf16*)alloc((size_t)F2 * H3 * 2);
    __bf16* tgW2p = (__bf16*)alloc((size_t)H3 * DOUT * 2);

    auto cdiv = [](int a, int b) { return (a + b - 1) / b; };

    // ---------------- pre-pack all weight matrices into B-fragment order
    for (int l = 0; l < 3; ++l) {
        pack_b<<<cdiv(D * F1, 256), 256, 0, stream>>>(
            W1 + (size_t)l * D * F1, W1p + (size_t)l * D * F1, D, F1);
        pack_b<<<cdiv(3 * F1 * F2, 256), 256, 0, stream>>>(
            W2 + (size_t)l * 3 * F1 * F2, W2p + (size_t)l * 3 * F1 * F2, 3 * F1, F2);
    }
    pack_b<<<cdiv(F2 * H3, 256), 256, 0, stream>>>(tfW1, tfW1p, F2, H3);
    pack_b<<<cdiv(H3 * DOUT, 256), 256, 0, stream>>>(tfW2, tfW2p, H3, DOUT);
    pack_b<<<cdiv(F2 * H3, 256), 256, 0, stream>>>(tgW1, tgW1p, F2, H3);
    pack_b<<<cdiv(H3 * DOUT, 256), 256, 0, stream>>>(tgW2, tgW2p, H3, DOUT);

    // ---------------- layer 1: h_l = x @ W1[l]; s1/s2; pack h for the P@h GEMM
    for (int l = 0; l < 3; ++l) {
        wmma_gemm<0><<<dim3(N / 16, F1 / 16), 32, 0, stream>>>(
            x, W1p + (size_t)l * D * F1, nullptr,
            h1 + (size_t)l * N * F1, N, F1, D, 0.f);
        row_dots<<<N, 32, 0, stream>>>(h1 + (size_t)l * N * F1, a1 + (size_t)l * 2 * F1,
                                       F1, s1b + (size_t)l * N, s2b + (size_t)l * N);
        pack_b<<<cdiv(N * F1, 256), 256, 0, stream>>>(
            h1 + (size_t)l * N * F1, h1p + (size_t)l * N * F1, N, F1);
    }
    row_neigh_max<<<N, 256, 0, stream>>>(adj, s2b, N, mb);
    for (int l = 0; l < 3; ++l) {
        flash_head<128><<<N / 16, 128, 0, stream>>>(
            s1b + (size_t)l * N, s2b + (size_t)l * N, mb + (size_t)l * N,
            adj, h1p + (size_t)l * N * F1, b1 + (size_t)l * F1,
            o1 + (size_t)l * N * F1, N);
    }
    elu_concat<<<cdiv(N * F1 * 3, 256), 256, 0, stream>>>(o1, hcat, N, F1);

    // ---------------- layer 2
    for (int l = 0; l < 3; ++l) {
        wmma_gemm<0><<<dim3(N / 16, F2 / 16), 32, 0, stream>>>(
            hcat, W2p + (size_t)l * 3 * F1 * F2, nullptr,
            h2 + (size_t)l * N * F2, N, F2, 3 * F1, 0.f);
        row_dots<<<N, 32, 0, stream>>>(h2 + (size_t)l * N * F2, a2 + (size_t)l * 2 * F2,
                                       F2, s1b2 + (size_t)l * N, s2b2 + (size_t)l * N);
        pack_b<<<cdiv(N * F2, 256), 256, 0, stream>>>(
            h2 + (size_t)l * N * F2, h2p + (size_t)l * N * F2, N, F2);
    }
    row_neigh_max<<<N, 256, 0, stream>>>(adj, s2b2, N, mb2);
    for (int l = 0; l < 3; ++l) {
        flash_head<64><<<N / 16, 128, 0, stream>>>(
            s1b2 + (size_t)l * N, s2b2 + (size_t)l * N, mb2 + (size_t)l * N,
            adj, h2p + (size_t)l * N * F2, b2 + (size_t)l * F2,
            o2 + (size_t)l * N * F2, N);
    }
    mean3<<<cdiv(N * F2, 256), 256, 0, stream>>>(o2, embed_out, N * F2);

    // ---------------- decoder MLPs (leaky 0.01) + pair dot
    wmma_gemm<1><<<dim3(N / 16, H3 / 16), 32, 0, stream>>>(
        embed_out, tfW1p, tfb1, tf1, N, H3, F2, SLOPE);
    wmma_gemm<1><<<dim3(N / 16, DOUT / 16), 32, 0, stream>>>(
        tf1, tfW2p, tfb2, tf_out, N, DOUT, H3, SLOPE);
    wmma_gemm<1><<<dim3(N / 16, H3 / 16), 32, 0, stream>>>(
        embed_out, tgW1p, tgb1, tg1, N, H3, F2, SLOPE);
    wmma_gemm<1><<<dim3(N / 16, DOUT / 16), 32, 0, stream>>>(
        tg1, tgW2p, tgb2, tg_out, N, DOUT, H3, SLOPE);

    pred_dot<<<cdiv(E, 256), 256, 0, stream>>>(ts, tf_out, tg_out, pred_out, E);
}
